// LSTM_Cell_28698971472164
// MI455X (gfx1250) — matrix operand order repaired and checked by reading, not compiled, not run on hardware
//
#include <hip/hip_runtime.h>

typedef __attribute__((ext_vector_type(16))) __bf16 v16bf;
typedef __attribute__((ext_vector_type(8)))  __bf16 v8bf;
typedef __attribute__((ext_vector_type(8)))  float   v8f;
typedef __attribute__((ext_vector_type(4))) unsigned int u32x4;
typedef __attribute__((ext_vector_type(8))) int          i32x8;
typedef __attribute__((ext_vector_type(4))) int          i32x4;

#define UNITS 1024
#define BATCH 4096
#define CDIM  2048

#define BM 128
#define BN 128
#define KSTEP 64                 // K staged per LDS buffer (2 WMMA sub-steps)
#define NST (CDIM / KSTEP)       // 32 stages
#define LDK2 72                  // padded row stride: 144B (16B-aligned, bank-spread)

struct __align__(8) bf4 { __bf16 x, y, z, w; };

__device__ __forceinline__ float sigf(float x) { return 1.0f / (1.0f + __expf(-x)); }

// LDS byte offset of a generic pointer into shared (flat addr[31:0] = LDS offset)
__device__ __forceinline__ unsigned lds_off(const void* p) {
  return (unsigned)(size_t)p;
}

// ---------------------------------------------------------------------------
// Prep 1: x = concat(hidden, inputs) -> bf16 row-major Xt[4096][2048]
// ---------------------------------------------------------------------------
__global__ __launch_bounds__(256)
void prep_x(const float* __restrict__ inp, const float* __restrict__ hid,
            __bf16* __restrict__ Xt)
{
  const size_t idx = ((size_t)blockIdx.x * 256 + threadIdx.x) * 8;
  const int m = (int)(idx >> 11);
  const int k = (int)(idx & 2047);
  const float* src = (k < 1024) ? (hid + (size_t)m * 1024 + k)
                                : (inp + (size_t)m * 1024 + (k - 1024));
  const float4 a = *(const float4*)src;
  const float4 b = *(const float4*)(src + 4);
  v8bf o;
  o[0] = (__bf16)a.x; o[1] = (__bf16)a.y; o[2] = (__bf16)a.z; o[3] = (__bf16)a.w;
  o[4] = (__bf16)b.x; o[5] = (__bf16)b.y; o[6] = (__bf16)b.z; o[7] = (__bf16)b.w;
  *(v8bf*)(Xt + idx) = o;
}

// ---------------------------------------------------------------------------
// Prep 2: W[g] (f32 [2048 x 1024], k x n) -> Wt bf16 packed transposed:
// Wt[g][st][n][k'] , k' = 0..63, contiguous 128B rows.
// ---------------------------------------------------------------------------
__global__ __launch_bounds__(256)
void prep_weights(const float* __restrict__ Wf, const float* __restrict__ Wi,
                  const float* __restrict__ Wc, const float* __restrict__ Wo,
                  __bf16* __restrict__ Wt)
{
  const int bx = blockIdx.x;                    // 0..511
  const int g  = bx >> 7;
  const float* W = (g == 0) ? Wf : (g == 1) ? Wi : (g == 2) ? Wc : Wo;
  const int rowid = (bx & 127) * 256 + threadIdx.x;  // 0..32767 = st*1024 + n
  const int st = rowid >> 10;
  const int n  = rowid & 1023;

  __bf16 row[KSTEP];
#pragma unroll
  for (int k = 0; k < KSTEP; ++k)               // lanes: consecutive n -> coalesced
    row[k] = (__bf16)W[(size_t)(st * KSTEP + k) * UNITS + n];

  __bf16* dst = Wt + ((size_t)(g * NST + st) * 1024 + n) * KSTEP;
#pragma unroll
  for (int q = 0; q < 8; ++q)
    *(v8bf*)(dst + q * 8) = *(const v8bf*)(row + q * 8);
}

// ---------------------------------------------------------------------------
// TDM: load a 128-row x 64-elem bf16 tile (rows strideElems apart in memory)
// into LDS, inserting 16B padding after every 128B row => LDS stride 144B.
// ---------------------------------------------------------------------------
__device__ __forceinline__ void tdm_tile_load(const __bf16* gsrc, unsigned ldsDst,
                                              unsigned strideElems) {
  const unsigned long long ga = (unsigned long long)(size_t)gsrc;
  u32x4 g0;
  g0[0] = 1u;                                         // count=1, no gather
  g0[1] = ldsDst;                                     // lds_addr (bytes)
  g0[2] = (unsigned)(ga & 0xFFFFFFFFu);               // global_addr[31:0]
  g0[3] = (unsigned)((ga >> 32) & 0x01FFFFFFu)        // global_addr[56:32]
        | (2u << 30);                                 // type=2 (image)
  i32x8 g1;
  g1[0] = (int)((1u << 16)        // data_size = 2B
              | (1u << 20)        // pad_enable
              | (4u << 22)        // pad_interval: 32 DWORDs (= 128B row)
              | (3u << 25));      // pad_amount:   4 DWORDs (= 16B pad)
  g1[1] = (int)(64u  << 16);      // tensor_dim0 = 64   (bits 79:48)
  g1[2] = (int)(128u << 16);      // tensor_dim1 = 128  (bits 111:80)
  g1[3] = (int)(64u  << 16);      // tile_dim0  = 64    (bits 127:112)
  g1[4] = 128;                    // tile_dim1  = 128   (bits 143:128)
  g1[5] = (int)strideElems;       // tensor_dim0_stride (bits 207:160, low word)
  g1[6] = 0;
  g1[7] = 0;
  i32x4 z4 = {0, 0, 0, 0};
  i32x8 z8 = {0, 0, 0, 0, 0, 0, 0, 0};
  __builtin_amdgcn_tensor_load_to_lds(g0, g1, z4, z4, z8, 0);
}

// ---------------------------------------------------------------------------
// Main: fused 4-gate GEMM (bf16 WMMA) + LSTM epilogue; all tiles TDM-fed.
// Dynamic LDS: Bls[2][4][128][72] then Als[2][128][72]  (184320 bytes)
// ---------------------------------------------------------------------------
#define SMEM_B_BYTES (2 * 4 * BN * LDK2 * 2)
#define SMEM_A_BYTES (2 * BM * LDK2 * 2)
#define SMEM_TOTAL   (SMEM_B_BYTES + SMEM_A_BYTES)

__global__ __launch_bounds__(256)
void lstm_cell_wmma_tdm(const __bf16* __restrict__ Xt,
                        const float* __restrict__ cell,
                        const __bf16* __restrict__ Wt,
                        const float* __restrict__ bfp, const float* __restrict__ bip,
                        const float* __restrict__ bcp, const float* __restrict__ bop,
                        float* __restrict__ outH, float* __restrict__ outC)
{
  extern __shared__ __align__(16) char smem[];
  __bf16 (*Bls)[4][BN][LDK2] = (__bf16 (*)[4][BN][LDK2])smem;             // [2][4][128][72]
  __bf16 (*Als)[BM][LDK2]    = (__bf16 (*)[BM][LDK2])(smem + SMEM_B_BYTES); // [2][128][72]

  const int tid  = threadIdx.x;
  const int lane = tid & 31;
  const int wv   = tid >> 5;
  const int m_off = (wv >> 2) * 64;
  const int n_off = (wv & 3) * 32;
  const int mblk = blockIdx.y * BM;
  const int nblk = blockIdx.x * BN;

  v8f acc[4][4][2];
#pragma unroll
  for (int g = 0; g < 4; ++g)
#pragma unroll
    for (int mi = 0; mi < 4; ++mi)
#pragma unroll
      for (int ni = 0; ni < 2; ++ni) {
        v8f z = {0.f,0.f,0.f,0.f,0.f,0.f,0.f,0.f};
        acc[g][mi][ni] = z;
      }

  // waves 0-3 each feed one gate's B tile; wave 4 feeds the A tile.
  auto issue_stage = [&](int buf, int st) {
    if (wv < 4) {
      const __bf16* src = Wt + ((size_t)(wv * NST + st) * 1024 + nblk) * KSTEP;
      tdm_tile_load(src, lds_off(&Bls[buf][wv][0][0]), KSTEP);     // packed rows
    } else if (wv == 4) {
      const __bf16* src = Xt + (size_t)mblk * CDIM + st * KSTEP;
      tdm_tile_load(src, lds_off(&Als[buf][0][0]), CDIM);          // strided rows
    }
  };

  const int lrow = lane & 15;
  const int hi   = lane >> 4;

  auto compute = [&](int buf) {
#pragma unroll
    for (int s = 0; s < 2; ++s) {            // two K=32 WMMA sub-steps per stage
      v16bf afrag[4];
#pragma unroll
      for (int mi = 0; mi < 4; ++mi) {
        const __bf16* p = &Als[buf][m_off + mi * 16 + lrow][s * 32 + hi * 8];
        v16bf a;
        *(v8bf*)&a       = *(const v8bf*)p;          // K 0-7 / 8-15
        *((v8bf*)&a + 1) = *(const v8bf*)(p + 16);   // K 16-23 / 24-31
        afrag[mi] = a;
      }
#pragma unroll
      for (int g = 0; g < 4; ++g) {
#pragma unroll
        for (int ni = 0; ni < 2; ++ni) {
          const __bf16* q = &Bls[buf][g][n_off + ni * 16 + lrow][s * 32 + hi * 8];
          v16bf b;
          *(v8bf*)&b       = *(const v8bf*)q;        // K 0-7 / 8-15
          *((v8bf*)&b + 1) = *(const v8bf*)(q + 16); // K 16-23 / 24-31
#pragma unroll
          for (int mi = 0; mi < 4; ++mi)
            acc[g][mi][ni] = __builtin_amdgcn_wmma_f32_16x16x32_bf16(
                false, afrag[mi], false, b, (short)0, acc[g][mi][ni], false, false);
        }
      }
    }
  };

  // ---- pipeline: TDM(st+1) overlaps WMMA(st) ----
  issue_stage(0, 0);
  if (wv < 5) __builtin_amdgcn_s_wait_tensorcnt(0);
  __syncthreads();

  for (int st = 0; st < NST; ++st) {
    const int buf = st & 1;
    if (st < NST - 1) issue_stage(buf ^ 1, st + 1);   // async into other buffer
    compute(buf);
    __syncthreads();                                  // reads of buf done
    if (st < NST - 1 && wv < 5) __builtin_amdgcn_s_wait_tensorcnt(0);
    __syncthreads();                                  // TDM data visible to all
  }

  // ---- fused LSTM epilogue (C/D layout: VGPR v -> row v / v+8, lane -> col) ----
#pragma unroll
  for (int ni = 0; ni < 2; ++ni) {
    const int n = nblk + n_off + ni * 16 + lrow;
    const float vbf = bfp[n], vbi = bip[n], vbc = bcp[n], vbo = bop[n];
#pragma unroll
    for (int mi = 0; mi < 4; ++mi) {
      const int mbase = mblk + m_off + mi * 16 + hi * 8;
#pragma unroll
      for (int v = 0; v < 8; ++v) {
        const int m = mbase + v;
        const size_t off = (size_t)m * UNITS + n;
        const float fg = sigf(acc[0][mi][ni][v] + vbf);
        const float ig = sigf(acc[1][mi][ni][v] + vbi);
        const float cg = tanhf(acc[2][mi][ni][v] + vbc);
        const float og = sigf(acc[3][mi][ni][v] + vbo);
        const float nc = fg * cell[off] + ig * cg;
        const float nh = og * tanhf(nc);
        outH[off] = nh;
        outC[off] = nc;
      }
    }
  }
}

// ---------------------------------------------------------------------------
// Fallback (workspace too small): self-contained kernel, VALU transpose, 51KB LDS.
// ---------------------------------------------------------------------------
#define BK 32
#define LDK 40
#define NKT (CDIM / BK)

__global__ __launch_bounds__(256)
void lstm_cell_wmma_fb(const float* __restrict__ inp,
                       const float* __restrict__ hid,
                       const float* __restrict__ cell,
                       const float* __restrict__ Wf, const float* __restrict__ bfp,
                       const float* __restrict__ Wi, const float* __restrict__ bip,
                       const float* __restrict__ Wc, const float* __restrict__ bcp,
                       const float* __restrict__ Wo, const float* __restrict__ bop,
                       float* __restrict__ outH, float* __restrict__ outC)
{
  __shared__ __align__(16) __bf16 Als[BM][LDK];
  __shared__ __align__(16) __bf16 Bls[4][BN][LDK];

  const int tid  = threadIdx.x;
  const int lane = tid & 31;
  const int wv   = tid >> 5;
  const int m_off = (wv >> 2) * 64;
  const int n_off = (wv & 3) * 32;
  const int mblk = blockIdx.y * BM;
  const int nblk = blockIdx.x * BN;
  const float* Wg[4] = { Wf, Wi, Wc, Wo };
  const int kq = tid & 7;
  const int nq = tid >> 3;

  v8f acc[4][4][2];
#pragma unroll
  for (int g = 0; g < 4; ++g)
#pragma unroll
    for (int mi = 0; mi < 4; ++mi)
#pragma unroll
      for (int ni = 0; ni < 2; ++ni) {
        v8f z = {0.f,0.f,0.f,0.f,0.f,0.f,0.f,0.f};
        acc[g][mi][ni] = z;
      }

  float4 areg[4];
  float4 breg[4][4];

  auto loadA = [&](int kt) {
    const float* src = (kt < 32) ? hid : inp;
    const int kk = (kt < 32) ? kt * BK : kt * BK - 1024;
#pragma unroll
    for (int i = 0; i < 4; ++i) {
      const int idx = tid + i * 256;
      const int row = idx >> 3;
      const int kqa = idx & 7;
      areg[i] = *(const float4*)(src + (size_t)(mblk + row) * 1024 + kk + kqa * 4);
    }
  };
  auto loadB = [&](int kt) {
#pragma unroll
    for (int g = 0; g < 4; ++g)
#pragma unroll
      for (int j = 0; j < 4; ++j)
        breg[g][j] = *(const float4*)(Wg[g] + (size_t)(kt * BK + kq * 4 + j) * UNITS
                                      + nblk + nq * 4);
  };
  auto stage = [&]() {
#pragma unroll
    for (int i = 0; i < 4; ++i) {
      const int idx = tid + i * 256;
      const int row = idx >> 3;
      const int kqa = idx & 7;
      const float* f = (const float*)&areg[i];
      bf4 v = { (__bf16)f[0], (__bf16)f[1], (__bf16)f[2], (__bf16)f[3] };
      *(bf4*)&Als[row][kqa * 4] = v;
    }
#pragma unroll
    for (int g = 0; g < 4; ++g) {
      const float* r0 = (const float*)&breg[g][0];
      const float* r1 = (const float*)&breg[g][1];
      const float* r2 = (const float*)&breg[g][2];
      const float* r3 = (const float*)&breg[g][3];
#pragma unroll
      for (int c = 0; c < 4; ++c) {
        bf4 v = { (__bf16)r0[c], (__bf16)r1[c], (__bf16)r2[c], (__bf16)r3[c] };
        *(bf4*)&Bls[g][nq * 4 + c][kq * 4] = v;
      }
    }
  };

  const int lrow = lane & 15;
  const int hi   = lane >> 4;
  auto compute = [&]() {
    v16bf afrag[4];
#pragma unroll
    for (int mi = 0; mi < 4; ++mi) {
      const __bf16* p = &Als[m_off + mi * 16 + lrow][hi * 8];
      v16bf a;
      *(v8bf*)&a       = *(const v8bf*)p;
      *((v8bf*)&a + 1) = *(const v8bf*)(p + 16);
      afrag[mi] = a;
    }
#pragma unroll
    for (int g = 0; g < 4; ++g) {
#pragma unroll
      for (int ni = 0; ni < 2; ++ni) {
        const __bf16* q = &Bls[g][n_off + ni * 16 + lrow][hi * 8];
        v16bf b;
        *(v8bf*)&b       = *(const v8bf*)q;
        *((v8bf*)&b + 1) = *(const v8bf*)(q + 16);
#pragma unroll
        for (int mi = 0; mi < 4; ++mi)
          acc[g][mi][ni] = __builtin_amdgcn_wmma_f32_16x16x32_bf16(
              false, afrag[mi], false, b, (short)0, acc[g][mi][ni], false, false);
      }
    }
  };

  loadA(0); loadB(0);
  for (int kt = 0; kt < NKT; ++kt) {
    stage();
    __syncthreads();
    if (kt < NKT - 1) { loadA(kt + 1); loadB(kt + 1); }
    compute();
    __syncthreads();
  }

#pragma unroll
  for (int ni = 0; ni < 2; ++ni) {
    const int n = nblk + n_off + ni * 16 + lrow;
    const float vbf = bfp[n], vbi = bip[n], vbc = bcp[n], vbo = bop[n];
#pragma unroll
    for (int mi = 0; mi < 4; ++mi) {
      const int mbase = mblk + m_off + mi * 16 + hi * 8;
#pragma unroll
      for (int v = 0; v < 8; ++v) {
        const int m = mbase + v;
        const size_t off = (size_t)m * UNITS + n;
        const float fg = sigf(acc[0][mi][ni][v] + vbf);
        const float ig = sigf(acc[1][mi][ni][v] + vbi);
        const float cg = tanhf(acc[2][mi][ni][v] + vbc);
        const float og = sigf(acc[3][mi][ni][v] + vbo);
        const float nc = fg * cell[off] + ig * cg;
        const float nh = og * tanhf(nc);
        outH[off] = nh;
        outC[off] = nc;
      }
    }
  }
}

extern "C" void kernel_launch(void* const* d_in, const int* in_sizes, int n_in,
                              void* d_out, int out_size, void* d_ws, size_t ws_size,
                              hipStream_t stream) {
  (void)in_sizes; (void)n_in; (void)out_size;
  const float* inp  = (const float*)d_in[0];
  const float* hid  = (const float*)d_in[1];
  const float* cell = (const float*)d_in[2];
  const float* Wf = (const float*)d_in[3];  const float* bf = (const float*)d_in[4];
  const float* Wi = (const float*)d_in[5];  const float* bi = (const float*)d_in[6];
  const float* Wc = (const float*)d_in[7];  const float* bc = (const float*)d_in[8];
  const float* Wo = (const float*)d_in[9];  const float* bo = (const float*)d_in[10];
  float* outH = (float*)d_out;
  float* outC = outH + (size_t)BATCH * UNITS;

  dim3 grid(UNITS / BN, BATCH / BM);        // (8, 32)
  const size_t wtBytes = (size_t)4 * CDIM * UNITS * sizeof(__bf16);      // 16 MiB
  const size_t xtBytes = (size_t)BATCH * CDIM * sizeof(__bf16);          // 16 MiB

  if (ws_size >= wtBytes + xtBytes) {
    __bf16* Wt = (__bf16*)d_ws;
    __bf16* Xt = (__bf16*)((char*)d_ws + wtBytes);
    prep_weights<<<512, 256, 0, stream>>>(Wf, Wi, Wc, Wo, Wt);
    prep_x<<<(BATCH * CDIM) / (256 * 8), 256, 0, stream>>>(inp, hid, Xt);
    (void)hipFuncSetAttribute((const void*)lstm_cell_wmma_tdm,
                              hipFuncAttributeMaxDynamicSharedMemorySize, SMEM_TOTAL);
    lstm_cell_wmma_tdm<<<grid, 256, SMEM_TOTAL, stream>>>(
        Xt, cell, Wt, bf, bi, bc, bo, outH, outC);
  } else {
    lstm_cell_wmma_fb<<<grid, 256, 0, stream>>>(
        inp, hid, cell, Wf, bf, Wi, bi, Wc, bc, Wo, bo, outH, outC);
  }
}
